// Transformer_14508399526756
// MI455X (gfx1250) — matrix operand order repaired and checked by reading, not compiled, not run on hardware
//
#include <hip/hip_runtime.h>
#include <hip/hip_bf16.h>

// ---------------------------------------------------------------------------
// Transformer encoder-decoder forward pass for MI455X (gfx1250, wave32).
// All large GEMMs + attention use v_wmma_f32_16x16x32_f16 (f16 in, f32 acc).
// fp32 -> fp16 conversion happens during global->LDS staging (TDM moves raw
// bytes only, so manual staging is the correct path). GEMM and attention use
// a 2-stage register->LDS double-buffered pipeline so global_load latency is
// hidden behind the WMMA work of the previous tile.
// ---------------------------------------------------------------------------

typedef _Float16 v16h __attribute__((ext_vector_type(16)));
typedef _Float16 v8h  __attribute__((ext_vector_type(8)));
typedef float    v8f  __attribute__((ext_vector_type(8)));

static constexpr int DM   = 1024;   // d_model
static constexpr int HH   = 16;     // heads
static constexpr int DKK  = 64;     // head dim
static constexpr int DFFC = 4096;   // ffn hidden
static constexpr int LAY  = 3;      // layers
static constexpr int VOC  = 16;
static constexpr int BB   = 4;
static constexpr int SL   = 1024;
static constexpr int MROWS = BB * SL;          // 4096

__device__ inline v8f wmma_f16(v16h a, v16h b, v8f c) {
  return __builtin_amdgcn_wmma_f32_16x16x32_f16(
      /*neg_a=*/false, a, /*neg_b=*/false, b,
      /*c_mod=*/(short)0, c, /*reuse_a=*/false, /*reuse_b=*/false);
}

// A-fragment (16x32, f16): lane m = lane&15, h = lane>>4;
// k = h*8 + [0..7]  and  16 + h*8 + [0..7]   (per ISA 16-bit A layout)
__device__ inline v16h load_frag_a(const _Float16* s, int ld, int lane) {
  int m = lane & 15, h = lane >> 4;
  const _Float16* p0 = s + m * ld + h * 8;
  const _Float16* p1 = s + m * ld + 16 + h * 8;
  v8h x0 = *(const v8h*)p0;
  v8h x1 = *(const v8h*)p1;
  v16h r;
#pragma unroll
  for (int i = 0; i < 8; ++i) { r[i] = x0[i]; r[8 + i] = x1[i]; }
  return r;
}

// B-fragment (32x16, f16) read from a TRANSPOSED tile Bt[n][k]:
// lane n = lane&15, h = lane>>4; k = h*8 + [0..7]  and  16 + h*8 + [0..7]
__device__ inline v16h load_frag_b(const _Float16* s, int ld, int lane) {
  int n = lane & 15, h = lane >> 4;
  const _Float16* p = s + n * ld + h * 8;
  v8h x0 = *(const v8h*)p;
  v8h x1 = *(const v8h*)(p + 16);
  v16h r;
#pragma unroll
  for (int i = 0; i < 8; ++i) { r[i] = x0[i]; r[8 + i] = x1[i]; }
  return r;
}

// ---------------------------------------------------------------------------
// GEMM: C[M,N] = act(A[M,K] * B[K,N] + bias (+ resid))   all fp32 in global.
// Block tile 128x128x32, 256 threads = 8 waves (2x4), wave tile 64x32.
// Double-buffered LDS: global loads for tile kt+1 are issued before the WMMA
// work on tile kt, hiding memory latency behind matrix math.
// ---------------------------------------------------------------------------
#define GBM 128
#define GBN 128
#define GBK 32
#define GLD 40   // LDS row stride in halves (80B, 16B aligned)

__global__ __launch_bounds__(256) void gemm_kernel(
    const float* __restrict__ A, const float* __restrict__ Bw,
    const float* __restrict__ bias, const float* __restrict__ resid,
    float* __restrict__ C, int M, int N, int K, int relu) {
  __shared__ _Float16 As[2][GBM * GLD];
  __shared__ _Float16 Bt[2][GBN * GLD];

  const int tid  = threadIdx.x;
  const int lane = tid & 31;
  const int wave = tid >> 5;
  const int wy = wave >> 2, wx = wave & 3;
  const int mBase = blockIdx.y * GBM;
  const int nBase = blockIdx.x * GBN;

  // Per-thread fixed staging assignment: A rows arow+32j, cols ac4..ac4+3;
  // B k-rows brow+8j, n-cols bc4..bc4+3 (stored transposed into Bt[n][k]).
  const int arow = tid >> 3, ac4 = (tid & 7) << 2;
  const int brow = tid >> 5, bc4 = (tid & 31) << 2;
  const float* aP = A + (size_t)(mBase + arow) * K + ac4;
  const float* bP = Bw + (size_t)brow * N + nBase + bc4;

  float4 rA[4], rB[4];
  auto load_tiles = [&](int kBase) {
#pragma unroll
    for (int j = 0; j < 4; ++j) {
      rA[j] = *(const float4*)(aP + (size_t)(32 * j) * K + kBase);
      rB[j] = *(const float4*)(bP + (size_t)(kBase + 8 * j) * N);
    }
  };
  auto store_tiles = [&](int sel) {
#pragma unroll
    for (int j = 0; j < 4; ++j) {
      _Float16* d = &As[sel][(arow + 32 * j) * GLD + ac4];
      d[0] = (_Float16)rA[j].x; d[1] = (_Float16)rA[j].y;
      d[2] = (_Float16)rA[j].z; d[3] = (_Float16)rA[j].w;
    }
#pragma unroll
    for (int j = 0; j < 4; ++j) {
      int kr = brow + 8 * j;
      Bt[sel][(bc4 + 0) * GLD + kr] = (_Float16)rB[j].x;
      Bt[sel][(bc4 + 1) * GLD + kr] = (_Float16)rB[j].y;
      Bt[sel][(bc4 + 2) * GLD + kr] = (_Float16)rB[j].z;
      Bt[sel][(bc4 + 3) * GLD + kr] = (_Float16)rB[j].w;
    }
  };

  const v8f z8 = {0.f, 0.f, 0.f, 0.f, 0.f, 0.f, 0.f, 0.f};
  v8f acc[4][2];
#pragma unroll
  for (int i = 0; i < 4; ++i)
#pragma unroll
    for (int j = 0; j < 2; ++j) acc[i][j] = z8;

  const int nk = K / GBK;
  load_tiles(0);
  store_tiles(0);
  __syncthreads();

  for (int kt = 0; kt < nk; ++kt) {
    const int sel = kt & 1;
    if (kt + 1 < nk) load_tiles((kt + 1) * GBK);  // issue early, wait later

    v16h af[4], bf[2];
#pragma unroll
    for (int i = 0; i < 4; ++i)
      af[i] = load_frag_a(&As[sel][(wy * 64 + i * 16) * GLD], GLD, lane);
#pragma unroll
    for (int j = 0; j < 2; ++j)
      bf[j] = load_frag_b(&Bt[sel][(wx * 32 + j * 16) * GLD], GLD, lane);
#pragma unroll
    for (int i = 0; i < 4; ++i)
#pragma unroll
      for (int j = 0; j < 2; ++j)
        acc[i][j] = wmma_f16(af[i], bf[j], acc[i][j]);

    if (kt + 1 < nk) store_tiles(sel ^ 1);
    __syncthreads();
  }

  // Epilogue: bias / residual / relu, C layout row = r + 8*h, col = lane&15
  const int h = lane >> 4, nl = lane & 15;
#pragma unroll
  for (int j = 0; j < 2; ++j) {
    int col = nBase + wx * 32 + j * 16 + nl;
    float bb = bias ? bias[col] : 0.f;
#pragma unroll
    for (int i = 0; i < 4; ++i) {
#pragma unroll
      for (int r = 0; r < 8; ++r) {
        int row = mBase + wy * 64 + i * 16 + r + 8 * h;
        float v = acc[i][j][r] + bb;
        if (resid) v += resid[(size_t)row * N + col];
        if (relu)  v = fmaxf(v, 0.f);
        C[(size_t)row * N + col] = v;
      }
    }
  }
}

// ---------------------------------------------------------------------------
// Flash-style attention: per (b, h, 64-query block). 128 thr = 4 waves,
// each wave owns 16 query rows. Key chunks of 32, online softmax, K/V chunk
// staging double-buffered like the GEMM.
// causal==0: key-padding mask from tok;  causal==1: (tok[q]!=0) && k<=q.
// ---------------------------------------------------------------------------
__global__ __launch_bounds__(128) void attn_kernel(
    const float* __restrict__ Q, const float* __restrict__ K,
    const float* __restrict__ V, float* __restrict__ O,
    const int* __restrict__ tok, int causal) {
  __shared__ _Float16 Qs[64 * 72];
  __shared__ _Float16 Ks[2][32 * 72];
  __shared__ _Float16 Vt[2][64 * 40];     // V transposed: Vt[dk][key]
  __shared__ _Float16 Ps[4 * 16 * 40];    // per-wave probs tile 16x32

  const int b   = blockIdx.z;
  const int hh  = blockIdx.y;
  const int qb0 = blockIdx.x * 64;
  const int tid = threadIdx.x, lane = tid & 31, wave = tid >> 5;
  const int h = lane >> 4, nl = lane & 15;
  const size_t base = ((size_t)b * SL) * DM + (size_t)hh * DKK;

  // Stage Q block (64 x 64) once
  for (int i = tid; i < 64 * 16; i += 128) {
    int row = i >> 4, c4 = (i & 15) << 2;
    float4 g = *(const float4*)(Q + base + (size_t)(qb0 + row) * DM + c4);
    _Float16* d = &Qs[row * 72 + c4];
    d[0] = (_Float16)g.x; d[1] = (_Float16)g.y;
    d[2] = (_Float16)g.z; d[3] = (_Float16)g.w;
  }
  __syncthreads();

  v16h a0 = load_frag_a(&Qs[wave * 16 * 72], 72, lane);       // dk 0..31
  v16h a1 = load_frag_a(&Qs[wave * 16 * 72 + 32], 72, lane);  // dk 32..63

  // Double-buffered K/V chunk staging: thread handles rows krow+8j.
  const int krow = tid >> 4, kc4 = (tid & 15) << 2;
  const float* kP = K + base + (size_t)krow * DM + kc4;
  const float* vP = V + base + (size_t)krow * DM + kc4;
  float4 rK[4], rV[4];
  auto load_kv = [&](int kc) {
#pragma unroll
    for (int j = 0; j < 4; ++j) {
      rK[j] = *(const float4*)(kP + (size_t)(kc * 32 + 8 * j) * DM);
      rV[j] = *(const float4*)(vP + (size_t)(kc * 32 + 8 * j) * DM);
    }
  };
  auto store_kv = [&](int sel) {
#pragma unroll
    for (int j = 0; j < 4; ++j) {
      int row = krow + 8 * j;
      _Float16* d = &Ks[sel][row * 72 + kc4];
      d[0] = (_Float16)rK[j].x; d[1] = (_Float16)rK[j].y;
      d[2] = (_Float16)rK[j].z; d[3] = (_Float16)rK[j].w;
      Vt[sel][(kc4 + 0) * 40 + row] = (_Float16)rV[j].x;
      Vt[sel][(kc4 + 1) * 40 + row] = (_Float16)rV[j].y;
      Vt[sel][(kc4 + 2) * 40 + row] = (_Float16)rV[j].z;
      Vt[sel][(kc4 + 3) * 40 + row] = (_Float16)rV[j].w;
    }
  };

  const v8f z8 = {0.f, 0.f, 0.f, 0.f, 0.f, 0.f, 0.f, 0.f};
  v8f o[4];
  float mrow[8], lrow[8];
  int qrow[8], qv[8];
#pragma unroll
  for (int t = 0; t < 4; ++t) o[t] = z8;
#pragma unroll
  for (int r = 0; r < 8; ++r) {
    mrow[r] = -3.0e38f; lrow[r] = 0.f;
    qrow[r] = qb0 + wave * 16 + r + 8 * h;
    qv[r] = causal ? (tok[(size_t)b * SL + qrow[r]] != 0) : 1;
  }

  load_kv(0);
  store_kv(0);
  __syncthreads();

  const int NC = SL / 32;
  for (int kc = 0; kc < NC; ++kc) {
    const int sel = kc & 1;
    if (kc + 1 < NC) load_kv(kc + 1);  // issue early

    // key-padding validity for this chunk (L2-hot per-lane loads)
    int kv0 = 1, kv1 = 1;
    if (!causal) {
      kv0 = tok[(size_t)b * SL + kc * 32 + nl] != 0;
      kv1 = tok[(size_t)b * SL + kc * 32 + 16 + nl] != 0;
    }

    // Scores: two 16x16 tiles over 32 keys, K-dim = 64 in two WMMA steps
    v8f c0 = z8, c1 = z8;
    c0 = wmma_f16(a0, load_frag_b(&Ks[sel][0],            72, lane), c0);
    c0 = wmma_f16(a1, load_frag_b(&Ks[sel][32],           72, lane), c0);
    c1 = wmma_f16(a0, load_frag_b(&Ks[sel][16 * 72],      72, lane), c1);
    c1 = wmma_f16(a1, load_frag_b(&Ks[sel][16 * 72 + 32], 72, lane), c1);

    const int k0 = kc * 32 + nl, k1 = k0 + 16;
    _Float16* pw = &Ps[wave * 16 * 40];
#pragma unroll
    for (int r = 0; r < 8; ++r) {
      float s0 = c0[r] * 0.125f, s1 = c1[r] * 0.125f;
      int v0, v1;
      if (causal) { v0 = qv[r] && (k0 <= qrow[r]); v1 = qv[r] && (k1 <= qrow[r]); }
      else        { v0 = kv0; v1 = kv1; }
      s0 = v0 ? s0 : -1.0e9f;
      s1 = v1 ? s1 : -1.0e9f;
      float t = fmaxf(s0, s1);
#pragma unroll
      for (int off = 8; off > 0; off >>= 1) t = fmaxf(t, __shfl_xor(t, off, 32));
      float mn = fmaxf(mrow[r], t);
      float p0 = __expf(s0 - mn), p1 = __expf(s1 - mn);
      float rs = p0 + p1;
#pragma unroll
      for (int off = 8; off > 0; off >>= 1) rs += __shfl_xor(rs, off, 32);
      float f = __expf(mrow[r] - mn);
      lrow[r] = lrow[r] * f + rs;
      mrow[r] = mn;
#pragma unroll
      for (int t4 = 0; t4 < 4; ++t4) o[t4][r] *= f;
      _Float16* pd = pw + (r + 8 * h) * 40;
      pd[nl]      = (_Float16)p0;
      pd[16 + nl] = (_Float16)p1;
    }
    __syncthreads();  // Ps visible (conservative; also spaces LDS phases)

    // O += P (16x32) x V (32x64): 4 WMMAs over the dk tiles
    v16h pf = load_frag_a(&Ps[wave * 16 * 40], 40, lane);
#pragma unroll
    for (int t4 = 0; t4 < 4; ++t4) {
      v16h vf = load_frag_b(&Vt[sel][t4 * 16 * 40], 40, lane);
      o[t4] = wmma_f16(pf, vf, o[t4]);
    }

    if (kc + 1 < NC) store_kv(sel ^ 1);
    __syncthreads();
  }

  // Normalize and write out (row-major [B*S, D] with head offset)
#pragma unroll
  for (int t4 = 0; t4 < 4; ++t4) {
#pragma unroll
    for (int r = 0; r < 8; ++r) {
      int q = qrow[r];
      O[base + (size_t)q * DM + t4 * 16 + nl] = o[t4][r] / lrow[r];
    }
  }
}

// ---------------------------------------------------------------------------
// LayerNorm: one block (256 thr) per row of 1024.
// ---------------------------------------------------------------------------
__device__ inline float block_reduce_sum(float v, float* sm) {
#pragma unroll
  for (int off = 16; off > 0; off >>= 1) v += __shfl_xor(v, off, 32);
  __syncthreads();
  if ((threadIdx.x & 31) == 0) sm[threadIdx.x >> 5] = v;
  __syncthreads();
  if (threadIdx.x == 0) {
    float s = 0.f;
    for (int i = 0; i < 8; ++i) s += sm[i];
    sm[8] = s;
  }
  __syncthreads();
  return sm[8];
}

__global__ __launch_bounds__(256) void ln_kernel(
    const float* __restrict__ in, const float* __restrict__ g,
    const float* __restrict__ bta, float* __restrict__ out) {
  __shared__ float sm[9];
  const int row = blockIdx.x;
  const int c = threadIdx.x * 4;
  float4 x = *(const float4*)(in + (size_t)row * DM + c);
  float s = x.x + x.y + x.z + x.w;
  s = block_reduce_sum(s, sm);
  const float mean = s * (1.0f / DM);
  float dx = x.x - mean, dy = x.y - mean, dz = x.z - mean, dw = x.w - mean;
  float sq = dx * dx + dy * dy + dz * dz + dw * dw;
  sq = block_reduce_sum(sq, sm);
  const float rs = rsqrtf(sq * (1.0f / DM) + 1e-5f);
  float4 gg = *(const float4*)(g + c);
  float4 bb = *(const float4*)(bta + c);
  float4 o;
  o.x = dx * rs * gg.x + bb.x;
  o.y = dy * rs * gg.y + bb.y;
  o.z = dz * rs * gg.z + bb.z;
  o.w = dw * rs * gg.w + bb.w;
  *(float4*)(out + (size_t)row * DM + c) = o;
}

// Embedding + positional add: X[row, :] = emb[tok[row]] + pos[row % S]
__global__ void embed_kernel(const int* __restrict__ tok,
                             const float* __restrict__ emb,
                             const float* __restrict__ pos,
                             float* __restrict__ out) {
  int idx = blockIdx.x * 256 + threadIdx.x;     // over MROWS * (DM/4)
  int row = idx >> 8;
  int c4 = (idx & 255) << 2;
  int s = row & (SL - 1);
  int t = tok[row];
  float4 e = *(const float4*)(emb + (size_t)t * DM + c4);
  float4 p = *(const float4*)(pos + (size_t)s * DM + c4);
  float4 o; o.x = e.x + p.x; o.y = e.y + p.y; o.z = e.z + p.z; o.w = e.w + p.w;
  *(float4*)(out + (size_t)row * DM + c4) = o;
}

// Final projection: out[row, v] = X[row,:] . fcW[:,v] + fcb[v]   (N=16)
__global__ void fc_kernel(const float* __restrict__ X,
                          const float* __restrict__ Wf,
                          const float* __restrict__ bf,
                          float* __restrict__ out) {
  int id = blockIdx.x * 256 + threadIdx.x;  // 65536
  int row = id >> 4, v = id & 15;
  float acc = bf[v];
  const float* xr = X + (size_t)row * DM;
  for (int k = 0; k < DM; ++k) acc += xr[k] * Wf[k * VOC + v];
  out[id] = acc;
}

// ---------------------------------------------------------------------------
// Host orchestration
// ---------------------------------------------------------------------------
static void gemm(const float* A, const float* Bw, const float* bias,
                 const float* resid, float* C, int M, int N, int K, int relu,
                 hipStream_t st) {
  dim3 g(N / GBN, M / GBM);
  gemm_kernel<<<g, 256, 0, st>>>(A, Bw, bias, resid, C, M, N, K, relu);
}

struct AttnW { const float *Wq, *bq, *Wk, *bk, *Wv, *bv, *Wo, *bo; };

extern "C" void kernel_launch(void* const* d_in, const int* in_sizes, int n_in,
                              void* d_out, int out_size, void* d_ws,
                              size_t ws_size, hipStream_t stream) {
  // -------- input mapping (handles insertion-order and sorted-pytree) ------
  const float* P[64];
  const int *src, *tgt;
  int off;
  bool srcFirst = (n_in >= 2 && in_sizes[0] == BB * SL && in_sizes[1] == BB * SL);
  if (srcFirst) { src = (const int*)d_in[0]; tgt = (const int*)d_in[1]; off = 2; }
  else          { src = (const int*)d_in[n_in - 2]; tgt = (const int*)d_in[n_in - 1]; off = 0; }
  for (int i = 0; i < n_in - 2; ++i) P[i] = (const float*)d_in[off + i];

  const float *emb, *pos, *fcW, *fcb;
  AttnW ea, da, ca;  // encoder attn, decoder self, decoder cross
  const float *eW1, *eb1, *eW2, *eb2, *eg1, *ebe1, *eg2, *ebe2;
  const float *dW1, *db1, *dW2, *db2, *dg1, *dbe1, *dg2, *dbe2, *dg3, *dbe3;

  bool sorted = srcFirst ? (in_sizes[2] != VOC * DM) : true;
  if (!sorted) {
    // insertion order: emb,pos,fcW,fcb, enc{attn{Wq,bq,Wk,bk,Wv,bv,Wo,bo},
    //   ff{W1,b1,W2,b2}, g1,be1,g2,be2}, dec{self,cross,ff,g1,be1,g2,be2,g3,be3}
    emb = P[0]; pos = P[1]; fcW = P[2]; fcb = P[3];
    ea = {P[4], P[5], P[6], P[7], P[8], P[9], P[10], P[11]};
    eW1 = P[12]; eb1 = P[13]; eW2 = P[14]; eb2 = P[15];
    eg1 = P[16]; ebe1 = P[17]; eg2 = P[18]; ebe2 = P[19];
    da = {P[20], P[21], P[22], P[23], P[24], P[25], P[26], P[27]};
    ca = {P[28], P[29], P[30], P[31], P[32], P[33], P[34], P[35]};
    dW1 = P[36]; db1 = P[37]; dW2 = P[38]; db2 = P[39];
    dg1 = P[40]; dbe1 = P[41]; dg2 = P[42]; dbe2 = P[43]; dg3 = P[44]; dbe3 = P[45];
  } else {
    // jax pytree (alphabetical): dec{be1,be2,be3,cross{Wk,Wo,Wq,Wv,bk,bo,bq,bv},
    //  ff{W1,W2,b1,b2}, g1,g2,g3, self{...}}, emb, enc{attn{...},be1,be2,
    //  ff{W1,W2,b1,b2}, g1,g2}, fcW, fcb, pos
    dbe1 = P[0]; dbe2 = P[1]; dbe3 = P[2];
    ca = {/*Wq*/P[5], /*bq*/P[9], /*Wk*/P[3], /*bk*/P[7],
          /*Wv*/P[6], /*bv*/P[10], /*Wo*/P[4], /*bo*/P[8]};
    dW1 = P[11]; dW2 = P[12]; db1 = P[13]; db2 = P[14];
    dg1 = P[15]; dg2 = P[16]; dg3 = P[17];
    da = {P[20], P[24], P[18], P[22], P[21], P[25], P[19], P[23]};
    emb = P[26];
    ea = {P[29], P[33], P[27], P[31], P[30], P[34], P[28], P[32]};
    ebe1 = P[35]; ebe2 = P[36];
    eW1 = P[37]; eW2 = P[38]; eb1 = P[39]; eb2 = P[40];
    eg1 = P[41]; eg2 = P[42];
    fcW = P[43]; fcb = P[44]; pos = P[45];
  }

  // -------- workspace layout (7 x 16MB = 112MB) ---------------------------
  const size_t NBUF = (size_t)MROWS * DM;  // 4,194,304 floats
  float* ws = (float*)d_ws;
  float* X  = ws;
  float* Qb = ws + 1 * NBUF;
  float* Kb = ws + 2 * NBUF;
  float* Vb = ws + 3 * NBUF;
  float* Tb = ws + 4 * NBUF;
  float* Ab = ws + 5 * NBUF;
  float* EN = ws + 6 * NBUF;
  float* FF = Qb;  // B*S*DFF == 4*NBUF: spans Qb..Tb (free during FFN)

  const size_t oDD = (size_t)DM * DM, oD = DM;
  const size_t oFW = (size_t)DM * DFFC, oF = DFFC;
  const dim3 agrid(SL / 64, HH, BB);
  const int M = MROWS;

  // ===== Encoder =====
  embed_kernel<<<MROWS * (DM / 4) / 256, 256, 0, stream>>>(src, emb, pos, X);
  for (int l = 0; l < LAY; ++l) {
    size_t w = l * oDD, b = l * oD, w1 = l * oFW, b1 = l * oF;
    gemm(X, ea.Wq + w, ea.bq + b, nullptr, Qb, M, DM, DM, 0, stream);
    gemm(X, ea.Wk + w, ea.bk + b, nullptr, Kb, M, DM, DM, 0, stream);
    gemm(X, ea.Wv + w, ea.bv + b, nullptr, Vb, M, DM, DM, 0, stream);
    attn_kernel<<<agrid, 128, 0, stream>>>(Qb, Kb, Vb, Tb, src, 0);
    gemm(Tb, ea.Wo + w, ea.bo + b, X, Ab, M, DM, DM, 0, stream);
    ln_kernel<<<MROWS, 256, 0, stream>>>(Ab, eg1 + b, ebe1 + b, X);
    gemm(X, eW1 + w1, eb1 + b1, nullptr, FF, M, DFFC, DM, 1, stream);
    gemm(FF, eW2 + w1, eb2 + b, X, Ab, M, DM, DFFC, 0, stream);
    ln_kernel<<<MROWS, 256, 0, stream>>>(Ab, eg2 + b, ebe2 + b, X);
  }
  hipMemcpyAsync(EN, X, NBUF * sizeof(float), hipMemcpyDeviceToDevice, stream);

  // ===== Decoder =====
  embed_kernel<<<MROWS * (DM / 4) / 256, 256, 0, stream>>>(tgt, emb, pos, X);
  for (int l = 0; l < LAY; ++l) {
    size_t w = l * oDD, b = l * oD, w1 = l * oFW, b1 = l * oF;
    // masked self-attention
    gemm(X, da.Wq + w, da.bq + b, nullptr, Qb, M, DM, DM, 0, stream);
    gemm(X, da.Wk + w, da.bk + b, nullptr, Kb, M, DM, DM, 0, stream);
    gemm(X, da.Wv + w, da.bv + b, nullptr, Vb, M, DM, DM, 0, stream);
    attn_kernel<<<agrid, 128, 0, stream>>>(Qb, Kb, Vb, Tb, tgt, 1);
    gemm(Tb, da.Wo + w, da.bo + b, X, Ab, M, DM, DM, 0, stream);
    ln_kernel<<<MROWS, 256, 0, stream>>>(Ab, dg1 + b, dbe1 + b, X);
    // cross-attention (K,V from encoder output)
    gemm(X,  ca.Wq + w, ca.bq + b, nullptr, Qb, M, DM, DM, 0, stream);
    gemm(EN, ca.Wk + w, ca.bk + b, nullptr, Kb, M, DM, DM, 0, stream);
    gemm(EN, ca.Wv + w, ca.bv + b, nullptr, Vb, M, DM, DM, 0, stream);
    attn_kernel<<<agrid, 128, 0, stream>>>(Qb, Kb, Vb, Tb, src, 0);
    gemm(Tb, ca.Wo + w, ca.bo + b, X, Ab, M, DM, DM, 0, stream);
    ln_kernel<<<MROWS, 256, 0, stream>>>(Ab, dg2 + b, dbe2 + b, X);
    // FFN
    gemm(X, dW1 + w1, db1 + b1, nullptr, FF, M, DFFC, DM, 1, stream);
    gemm(FF, dW2 + w1, db2 + b, X, Ab, M, DM, DFFC, 0, stream);
    ln_kernel<<<MROWS, 256, 0, stream>>>(Ab, dg3 + b, dbe3 + b, X);
  }

  // Final projection to vocab
  fc_kernel<<<MROWS * VOC / 256, 256, 0, stream>>>(X, fcW, fcb, (float*)d_out);
}
